// SGC_2362232013426
// MI455X (gfx1250) — matrix operand order repaired and checked by reading, not compiled, not run on hardware
//
#include <hip/hip_runtime.h>

typedef __attribute__((ext_vector_type(16))) _Float16 v16h;
typedef __attribute__((ext_vector_type(8)))  float    v8f;

#define D_FEAT 64
#define C_OUT  40

// ---------------- gcn_norm ----------------

// deg starts at 1.0 (self-loop of weight 1 per node)
__global__ void k_deg_init(float* __restrict__ deg, int N) {
    int i = blockIdx.x * blockDim.x + threadIdx.x;
    if (i < N) deg[i] = 1.0f;
}

__global__ void k_deg_accum(const int* __restrict__ col, const float* __restrict__ w,
                            float* __restrict__ deg, int E) {
    int i = blockIdx.x * blockDim.x + threadIdx.x;
    if (i < E) atomicAdd(&deg[col[i]], w[i]);
}

// deg -> deg^{-1/2} in place (deg >= 1 always due to self loops)
__global__ void k_dinv(float* __restrict__ deg, int N) {
    int i = blockIdx.x * blockDim.x + threadIdx.x;
    if (i < N) {
        float d = deg[i];
        deg[i] = (d > 0.0f) ? rsqrtf(d) : 0.0f;
    }
}

// ---------------- SpMM layer ----------------

// xd[i,:] = dinv[i]^2 * xs[i,:]   (self-loop contribution; also zero-inits xd)
__global__ void k_spmm_init(const float* __restrict__ xs, const float* __restrict__ dinv,
                            float* __restrict__ xd, int nf) {
    int idx = blockIdx.x * blockDim.x + threadIdx.x;
    if (idx < nf) {
        int i = idx >> 6;           // node
        float s = dinv[i];
        xd[idx] = s * s * xs[idx];
    }
}

// edge-parallel scatter-add: one thread handles 2 consecutive features of one edge
__global__ void k_spmm_edge(const float* __restrict__ xs, const int* __restrict__ row,
                            const int* __restrict__ col, const float* __restrict__ w,
                            const float* __restrict__ dinv, float* __restrict__ xd, int E) {
    int idx = blockIdx.x * blockDim.x + threadIdx.x;
    int e = idx >> 5;               // edge
    int f = (idx & 31) << 1;        // feature pair
    if (e < E) {
        int r = row[e];
        int c = col[e];
        float nrm = dinv[r] * w[e] * dinv[c];
        const float2 v = *(const float2*)(xs + (size_t)r * D_FEAT + f);
        float* dst = xd + (size_t)c * D_FEAT + f;
        atomicAdd(dst,     nrm * v.x);
        atomicAdd(dst + 1, nrm * v.y);
    }
}

// ---------------- dense projection via WMMA ----------------
// out[N, 40] = X[N, 64] @ W[40, 64]^T + b
// One wave32 per 16-row tile; K=64 as two 16x16x32 f16 WMMAs; 3 column tiles (48, padded).
__global__ void k_gemm_wmma(const float* __restrict__ X, const float* __restrict__ Wt,
                            const float* __restrict__ bias, float* __restrict__ out,
                            int N) {
    int wave = (blockIdx.x * blockDim.x + threadIdx.x) >> 5;
    int lane = threadIdx.x & 31;
    int m0 = wave * 16;
    if (m0 >= N) return;            // wave-uniform: EXEC stays all-1s for live waves

    const bool hi  = lane >= 16;
    const int  l15 = lane & 15;

    // ---- A tiles: 16x32 f16 per k-step. Lane holds row m0+l15.
    // halves i<8  -> K = kb + i        (kb = 0 for lanes 0-15, 8 for lanes 16-31)
    // halves i>=8 -> K = 16 + kb + (i-8)
    v16h a0, a1;
    {
        int m = m0 + l15;
        if (m >= N) m = N - 1;      // safe clamp (N%16==0 in practice)
        const float* xr = X + (size_t)m * D_FEAT;
        const int kb = hi ? 8 : 0;
#pragma unroll
        for (int i = 0; i < 8; ++i) {
            a0[i]     = (_Float16)xr[kb + i];
            a0[8 + i] = (_Float16)xr[16 + kb + i];
            a1[i]     = (_Float16)xr[32 + kb + i];
            a1[8 + i] = (_Float16)xr[48 + kb + i];
        }
    }

    // ---- B tiles + accumulate. B[k][n] = W[n][k]; lane holds column n = nt*16 + l15,
    // halves i<8 -> K = (hi?8:0) + i, i>=8 -> K = 16 + (hi?8:0) + (i-8). Zero-pad n >= 40.
    v8f acc[3];
#pragma unroll
    for (int nt = 0; nt < 3; ++nt) {
        v16h b0, b1;
        const int n  = nt * 16 + l15;
        const int kb = hi ? 8 : 0;
        if (n < C_OUT) {
            const float* wr = Wt + (size_t)n * D_FEAT;
#pragma unroll
            for (int i = 0; i < 8; ++i) {
                b0[i]     = (_Float16)wr[kb + i];
                b0[8 + i] = (_Float16)wr[16 + kb + i];
                b1[i]     = (_Float16)wr[32 + kb + i];
                b1[8 + i] = (_Float16)wr[48 + kb + i];
            }
        } else {
#pragma unroll
            for (int i = 0; i < 16; ++i) { b0[i] = (_Float16)0.0f; b1[i] = (_Float16)0.0f; }
        }
        v8f c = {};
        c = __builtin_amdgcn_wmma_f32_16x16x32_f16(false, a0, false, b0, (short)0, c, false, false);
        c = __builtin_amdgcn_wmma_f32_16x16x32_f16(false, a1, false, b1, (short)0, c, false, false);
        acc[nt] = c;
    }

    // ---- Store + bias. VGPR r: lanes 0-15 -> M = m0+r, lanes 16-31 -> M = m0+8+r; N = nt*16+l15.
#pragma unroll
    for (int nt = 0; nt < 3; ++nt) {
        const int n = nt * 16 + l15;
        if (n < C_OUT) {
            const float bb = bias[n];
            const int mbase = m0 + (hi ? 8 : 0);
#pragma unroll
            for (int r = 0; r < 8; ++r) {
                const int m = mbase + r;
                if (m < N) out[(size_t)m * C_OUT + n] = acc[nt][r] + bb;
            }
        }
    }
}

// ---------------- driver ----------------

extern "C" void kernel_launch(void* const* d_in, const int* in_sizes, int n_in,
                              void* d_out, int out_size, void* d_ws, size_t ws_size,
                              hipStream_t stream) {
    const float* x    = (const float*)d_in[0];   // [N, 64]
    const int*   eidx = (const int*)  d_in[1];   // [2, E]
    const float* ew   = (const float*)d_in[2];   // [E]
    const float* W    = (const float*)d_in[3];   // [40, 64]
    const float* b    = (const float*)d_in[4];   // [40]
    float*       out  = (float*)d_out;           // [N, 40]

    const int N = in_sizes[0] / D_FEAT;
    const int E = in_sizes[2];
    const int* row = eidx;
    const int* col = eidx + E;

    // workspace layout: dinv[N] | xA[N*64] | xB[N*64]
    char* ws = (char*)d_ws;
    float* dinv = (float*)ws;
    size_t off = (((size_t)N * sizeof(float)) + 255) & ~(size_t)255;
    float* xA = (float*)(ws + off);
    float* xB = xA + (size_t)N * D_FEAT;

    const int T = 256;

    // gcn_norm
    k_deg_init <<<(N + T - 1) / T, T, 0, stream>>>(dinv, N);
    k_deg_accum<<<(E + T - 1) / T, T, 0, stream>>>(col, ew, dinv, E);
    k_dinv     <<<(N + T - 1) / T, T, 0, stream>>>(dinv, N);

    const int nf = N * D_FEAT;      // 6.4M
    const int et = E * 32;          // 51.2M thread slots (2 features each)

    // layer 1: x -> xA
    k_spmm_init<<<(nf + T - 1) / T, T, 0, stream>>>(x,  dinv, xA, nf);
    k_spmm_edge<<<(et + T - 1) / T, T, 0, stream>>>(x,  row, col, ew, dinv, xA, E);
    // layer 2: xA -> xB
    k_spmm_init<<<(nf + T - 1) / T, T, 0, stream>>>(xA, dinv, xB, nf);
    k_spmm_edge<<<(et + T - 1) / T, T, 0, stream>>>(xA, row, col, ew, dinv, xB, E);

    // dense projection with WMMA: one wave per 16-row tile
    const int tiles   = (N + 15) / 16;          // 6250
    const int threads = tiles * 32;
    k_gemm_wmma<<<(threads + T - 1) / T, T, 0, stream>>>(xB, W, b, out, N);
}
